// DetectionHead_38620345925752
// MI455X (gfx1250) — matrix operand order repaired and checked by reading, not compiled, not run on hardware
//
#include <hip/hip_runtime.h>
#include <hip/hip_bf16.h>

// ---------- WMMA / TDM vector types ----------
typedef _Float16 v16h  __attribute__((ext_vector_type(16)));
typedef _Float16 half8 __attribute__((ext_vector_type(8)));
typedef float    v8f   __attribute__((ext_vector_type(8)));
typedef unsigned int u32x4 __attribute__((ext_vector_type(4)));
typedef int          i32x4 __attribute__((ext_vector_type(4)));
typedef int          i32x8 __attribute__((ext_vector_type(8)));

#define KK 5          // spline kernel size per dim
#define NK 25         // K*K kernel matrices
#define CIN 32

// ---------- Tensor Data Mover: 1-D global->LDS DMA (D# per ISA 8.3/8.4) ----------
// Moves ndwords dwords from gptr to LDS byte-offset lds_off. Caller waits with
// s_wait_tensorcnt and a block barrier before consuming.
__device__ __forceinline__ void tdm_load_1d(unsigned lds_off, const void* gptr, unsigned ndwords) {
    unsigned long long ga = (unsigned long long)(size_t)gptr;
    u32x4 g0;
    g0[0] = 1u;                                   // count=1, user descriptor
    g0[1] = lds_off;                              // lds_addr (bytes)
    g0[2] = (unsigned)ga;                         // global_addr[31:0]
    g0[3] = (unsigned)(ga >> 32) | 0x80000000u;   // global_addr[56:32] | type=2 (bits 127:126)
    i32x8 g1;
    g1[0] = (2 << 16);                            // workgroup_mask=0, data_size=4B, no flags
    g1[1] = (int)((ndwords & 0xFFFFu) << 16);     // tensor_dim0[15:0]  (bits 63:48)
    g1[2] = (int)(((ndwords >> 16) & 0xFFFFu) | (1u << 16)); // dim0 hi | tensor_dim1=1
    g1[3] = (int)((ndwords & 0xFFFFu) << 16);     // tile_dim0 (bits 127:112)
    g1[4] = 0;                                    // tile_dim1=0, tile_dim2=0 (unused)
    g1[5] = (int)ndwords;                         // tensor_dim0_stride[31:0]
    g1[6] = 0;
    g1[7] = 0;
    i32x4 z4 = {0, 0, 0, 0};                      // groups 2/3 unused (<=2D tensor)
    i32x8 z8 = {0, 0, 0, 0, 0, 0, 0, 0};
    __builtin_amdgcn_tensor_load_to_lds(g0, g1, z4, z4, z8, 0);
}

// ---------- utility kernels ----------
__global__ __launch_bounds__(256) void k_zero(float* p, long n) {
    long t = (long)blockIdx.x * blockDim.x + threadIdx.x;
    if (t < n) p[t] = 0.0f;
}

__global__ __launch_bounds__(256) void k_to_half(const float* __restrict__ x,
                                                 _Float16* __restrict__ xh, long n) {
    long t = (long)blockIdx.x * blockDim.x + threadIdx.x;
    if (t < n) xh[t] = (_Float16)x[t];
}

// W [25][32][32] (k,i,o) fp32  ->  Wt [25][32][32] (k,o,i) f16  (transposed, i contiguous)
__global__ __launch_bounds__(256) void k_prep_w(const float* __restrict__ W,
                                                _Float16* __restrict__ Wt) {
    int t = blockIdx.x * blockDim.x + threadIdx.x;
    if (t >= NK * CIN * CIN) return;
    int k = t / (CIN * CIN);
    int rem = t - k * (CIN * CIN);
    int o = rem >> 5, i = rem & 31;
    Wt[t] = (_Float16)W[(k * CIN + i) * CIN + o];
}

__global__ __launch_bounds__(256) void k_degree(const int* __restrict__ ei, float* __restrict__ deg, int E) {
    int e = blockIdx.x * blockDim.x + threadIdx.x;
    if (e < E) atomicAdd(deg + ei[E + e], 1.0f);
}

// ---------- WMMA weight-expansion GEMM: xWh[n][k][o] = sum_i xh[n][i] * W[k][i][o] ----------
// One wave per (16-node tile, 16-col tile); A loaded once, 25 WMMAs over k with B in LDS.
// All 25 W_k^T tiles (50 KB) staged into LDS by one TDM DMA per block.
__global__ __launch_bounds__(256) void k_xw_wmma(const _Float16* __restrict__ xh,
                                                 const _Float16* __restrict__ Wt,
                                                 _Float16* __restrict__ xWh,
                                                 int ntiles) {
    __shared__ _Float16 Bs[NK * CIN * CIN];       // 50 KB
    if (threadIdx.x < 32) {                       // wave 0 issues the tensor DMA
        tdm_load_1d((unsigned)(size_t)Bs, Wt, (NK * CIN * CIN * 2) / 4);
        __builtin_amdgcn_s_wait_tensorcnt(0);
    }
    __syncthreads();

    int wave = blockIdx.x * (blockDim.x >> 5) + (threadIdx.x >> 5);
    int lane = threadIdx.x & 31;
    int ct   = wave & 1;          // column tile (cout 0..15 / 16..31)
    int tile = wave >> 1;         // 16-node tile
    if (tile < ntiles) {          // wave-uniform: EXEC all-ones for WMMA
        int r = lane & 15, h = lane >> 4;

        // A (16x32 f16): lane group h=0 holds K 0..15; h=1 holds K 16..31
        const half8* ap = (const half8*)(xh + ((size_t)(tile * 16 + r)) * CIN + h * 16);
        half8 alo = ap[0];
        half8 ahi = ap[1];        // +8 halfs
        v16h a;
#pragma unroll
        for (int i = 0; i < 8; ++i) { a[i] = alo[i]; a[8 + i] = ahi[i]; }

        // B (32x16 f16) from LDS: lane = column, lane group selects K 0..15 vs 16..31
        const _Float16* bbase = Bs + (ct * 16 + r) * CIN + h * 16;
        _Float16* dbase = xWh + (size_t)(tile * 16) * (NK * CIN) + ct * 16 + r;

#pragma unroll 5
        for (int k = 0; k < NK; ++k) {
            v16h b = *(const v16h*)(bbase + k * (CIN * CIN));
            v8f c = {};
            c = __builtin_amdgcn_wmma_f32_16x16x32_f16(false, a, false, b, (short)0, c, false, false);
            // D 16x16 f32: lane holds column r, VGPR v -> row v + 8*h
            _Float16* dp = dbase + k * CIN;
#pragma unroll
            for (int v = 0; v < 8; ++v)
                dp[(size_t)(v + 8 * h) * (NK * CIN)] = (_Float16)c[v];
        }
    }
}

// ---------- edge message + segment-sum: one wave per edge, lane == channel ----------
__global__ __launch_bounds__(256) void k_edge_msg(const _Float16* __restrict__ xWh,
                                                  const int* __restrict__ ei,
                                                  const float* __restrict__ ea,
                                                  float* __restrict__ agg, int E) {
    int e    = blockIdx.x * (blockDim.x >> 5) + (threadIdx.x >> 5);
    int lane = threadIdx.x & 31;
    if (e >= E) return;
    __builtin_prefetch(ea + 2 * e + 1024, 0, 1);   // global_prefetch_b8

    int src = ei[e];
    int dst = ei[E + e];
    float px = ea[2 * e]     * (float)(KK - 1);
    float py = ea[2 * e + 1] * (float)(KK - 1);
    float fx0 = fminf(fmaxf(floorf(px), 0.0f), (float)(KK - 2));
    float fy0 = fminf(fmaxf(floorf(py), 0.0f), (float)(KK - 2));
    float fx = px - fx0, fy = py - fy0;
    int k00 = (int)fx0 + (int)fy0 * KK;

    float w00 = (1.0f - fx) * (1.0f - fy);
    float w10 = fx * (1.0f - fy);
    float w01 = (1.0f - fx) * fy;
    float w11 = fx * fy;

    const _Float16* base = xWh + (size_t)src * (NK * CIN) + lane;
    float m = 0.0f;
    m += w00 * (float)base[(size_t)(k00)          * CIN];
    m += w10 * (float)base[(size_t)(k00 + 1)      * CIN];
    m += w01 * (float)base[(size_t)(k00 + KK)     * CIN];
    m += w11 * (float)base[(size_t)(k00 + KK + 1) * CIN];
    atomicAdd(agg + (size_t)dst * CIN + lane, m);
}

// ---------- finalize conv: y = agg/max(deg,1) + x@root ; accumulate BN stats ----------
__global__ __launch_bounds__(256) void k_finalize(const float* __restrict__ agg,
                                                  const float* __restrict__ deg,
                                                  const float* __restrict__ xin,
                                                  const float* __restrict__ root,
                                                  float* __restrict__ y,
                                                  float* __restrict__ stats, int N) {
    __shared__ float Rs[CIN * CIN];
    __shared__ float ls[64];
    for (int i = threadIdx.x; i < CIN * CIN; i += blockDim.x) Rs[i] = root[i];
    if (threadIdx.x < 64) ls[threadIdx.x] = 0.0f;
    __syncthreads();

    long t = (long)blockIdx.x * blockDim.x + threadIdx.x;  // t = n*32 + c
    if (t < (long)N * CIN) {
        int n = (int)(t >> 5), c = (int)(t & 31);
        float d = fmaxf(deg[n], 1.0f);
        float s = agg[t] / d;
        const float* xr = xin + (size_t)n * CIN;
        float r = 0.0f;
#pragma unroll
        for (int i = 0; i < CIN; ++i) r += xr[i] * Rs[i * CIN + c];
        float val = s + r;
        y[t] = val;
        atomicAdd(&ls[c], val);            // ds_add_f32
        atomicAdd(&ls[32 + c], val * val);
    }
    __syncthreads();
    if (threadIdx.x < 64) atomicAdd(&stats[threadIdx.x], ls[threadIdx.x]);
}

__global__ __launch_bounds__(256) void k_bn_apply(float* __restrict__ y,
                                                  const float* __restrict__ stats,
                                                  const float* __restrict__ g,
                                                  const float* __restrict__ beta, int N) {
    long t = (long)blockIdx.x * blockDim.x + threadIdx.x;
    if (t >= (long)N * CIN) return;
    int c = (int)(t & 31);
    float invN = 1.0f / (float)N;
    float m = stats[c] * invN;
    float v = stats[32 + c] * invN - m * m;
    float o = (y[t] - m) * rsqrtf(v + 1e-5f) * g[c] + beta[c];
    y[t] = fmaxf(o, 0.0f);
}

// ---------- head conv edge pass (cout small): thread per edge, W staged by TDM ----------
template <int COUT>
__global__ __launch_bounds__(256) void k_head_edge(const float* __restrict__ x,
                                                   const float* __restrict__ W,
                                                   const int* __restrict__ ei,
                                                   const float* __restrict__ ea,
                                                   float* __restrict__ agg, int E) {
    __shared__ float Ws[NK * CIN * COUT];
    if (threadIdx.x < 32) {
        tdm_load_1d((unsigned)(size_t)Ws, W, NK * CIN * COUT);
        __builtin_amdgcn_s_wait_tensorcnt(0);
    }
    __syncthreads();

    int e = blockIdx.x * blockDim.x + threadIdx.x;
    if (e >= E) return;
    __builtin_prefetch(ea + 2 * e + 2048, 0, 1);

    int src = ei[e];
    int dst = ei[E + e];
    float px = ea[2 * e]     * (float)(KK - 1);
    float py = ea[2 * e + 1] * (float)(KK - 1);
    float fx0 = fminf(fmaxf(floorf(px), 0.0f), (float)(KK - 2));
    float fy0 = fminf(fmaxf(floorf(py), 0.0f), (float)(KK - 2));
    float fx = px - fx0, fy = py - fy0;
    int k00 = (int)fx0 + (int)fy0 * KK;

    int   ki[4] = { k00, k00 + 1, k00 + KK, k00 + KK + 1 };
    float wt[4] = { (1.0f - fx) * (1.0f - fy), fx * (1.0f - fy),
                    (1.0f - fx) * fy,           fx * fy };

    float xv[CIN];
    const float* xr = x + (size_t)src * CIN;
#pragma unroll
    for (int i = 0; i < CIN; ++i) xv[i] = xr[i];

    float acc[COUT];
#pragma unroll
    for (int o = 0; o < COUT; ++o) acc[o] = 0.0f;

#pragma unroll
    for (int t = 0; t < 4; ++t) {
        const float* wk = Ws + ki[t] * (CIN * COUT);
#pragma unroll
        for (int o = 0; o < COUT; ++o) {
            float s = 0.0f;
#pragma unroll
            for (int i = 0; i < CIN; ++i) s += xv[i] * wk[i * COUT + o];
            acc[o] += wt[t] * s;
        }
    }
#pragma unroll
    for (int o = 0; o < COUT; ++o)
        atomicAdd(agg + (size_t)dst * COUT + o, acc[o]);
}

// ---------- head finalize + grid scatter ----------
__global__ __launch_bounds__(256) void k_head_scatter(const float* __restrict__ aggR,
                                                      const float* __restrict__ aggC,
                                                      const float* __restrict__ aggO,
                                                      const float* __restrict__ deg,
                                                      const float* __restrict__ x2,
                                                      const float* __restrict__ x3,
                                                      const float* __restrict__ rootr, const float* __restrict__ br,
                                                      const float* __restrict__ rootc, const float* __restrict__ bc,
                                                      const float* __restrict__ rooto, const float* __restrict__ bo,
                                                      const float* __restrict__ pos,
                                                      const int*  __restrict__ batch,
                                                      float* __restrict__ out, int N) {
    __shared__ float Rr[CIN * 4], Rc[CIN * 2], Ro[CIN];
    for (int i = threadIdx.x; i < CIN * 4; i += blockDim.x) Rr[i] = rootr[i];
    for (int i = threadIdx.x; i < CIN * 2; i += blockDim.x) Rc[i] = rootc[i];
    for (int i = threadIdx.x; i < CIN;     i += blockDim.x) Ro[i] = rooto[i];
    __syncthreads();

    int n = blockIdx.x * blockDim.x + threadIdx.x;
    if (n >= N) return;

    float d = fmaxf(deg[n], 1.0f);
    const float* xr2 = x2 + (size_t)n * CIN;
    const float* xr3 = x3 + (size_t)n * CIN;

    float reg[4];
#pragma unroll
    for (int o = 0; o < 4; ++o) {
        float r = 0.0f;
#pragma unroll
        for (int i = 0; i < CIN; ++i) r += xr2[i] * Rr[i * 4 + o];
        reg[o] = aggR[(size_t)n * 4 + o] / d + r + br[o];
    }
    float cls[2];
#pragma unroll
    for (int o = 0; o < 2; ++o) {
        float r = 0.0f;
#pragma unroll
        for (int i = 0; i < CIN; ++i) r += xr3[i] * Rc[i * 2 + o];
        cls[o] = aggC[(size_t)n * 2 + o] / d + r + bc[o];
    }
    float ro = 0.0f;
#pragma unroll
    for (int i = 0; i < CIN; ++i) ro += xr3[i] * Ro[i];
    float obj = aggO[n] / d + ro + bo[0];

    int gx = (int)floorf(pos[2 * n]     * 16.0f);  // WIDTH/STRIDE  = 240/15
    int gy = (int)floorf(pos[2 * n + 1] * 12.0f);  // HEIGHT/STRIDE = 180/15
    gx = min(max(gx, 0), 15);
    gy = min(max(gy, 0), 11);
    int b = batch[n];
    int cell = gy * 16 + gx;

    float* co = out;                 // cls  [8,2,12,16] = 3072
    float* rg = out + 3072;          // reg  [8,4,12,16] = 6144
    float* ob = out + 3072 + 6144;   // obj  [8,1,12,16] = 1536
    co[(b * 2 + 0) * 192 + cell] = cls[0];
    co[(b * 2 + 1) * 192 + cell] = cls[1];
#pragma unroll
    for (int o = 0; o < 4; ++o) rg[(b * 4 + o) * 192 + cell] = reg[o];
    ob[b * 192 + cell] = obj;
}

// ---------- host orchestration ----------
extern "C" void kernel_launch(void* const* d_in, const int* in_sizes, int n_in,
                              void* d_out, int out_size, void* d_ws, size_t ws_size,
                              hipStream_t stream) {
    const int N = in_sizes[0] / CIN;
    const int E = in_sizes[1] / 2;
    const int ntiles = (N + 15) / 16;
    const long NP = (long)ntiles * 16;        // padded node count

    const float* x     = (const float*)d_in[0];
    const int*   ei    = (const int*)  d_in[1];
    const float* ea    = (const float*)d_in[2];
    const float* pos   = (const float*)d_in[3];
    const int*   batch = (const int*)  d_in[4];
    const float* W1 = (const float*)d_in[5],  *root1 = (const float*)d_in[6];
    const float* W2 = (const float*)d_in[7],  *root2 = (const float*)d_in[8];
    const float* W3 = (const float*)d_in[9],  *root3 = (const float*)d_in[10];
    const float* Wr = (const float*)d_in[11], *rootr = (const float*)d_in[12], *br = (const float*)d_in[13];
    const float* Wc = (const float*)d_in[14], *rootc = (const float*)d_in[15], *bc = (const float*)d_in[16];
    const float* Wo = (const float*)d_in[17], *rooto = (const float*)d_in[18], *bo = (const float*)d_in[19];
    const float* g1 = (const float*)d_in[20], *beta1 = (const float*)d_in[21];
    const float* g2 = (const float*)d_in[22], *beta2 = (const float*)d_in[23];
    const float* g3 = (const float*)d_in[24], *beta3 = (const float*)d_in[25];

    // workspace carve
    char* wsp = (char*)d_ws;
    size_t off = 0;
    auto carve = [&](size_t bytes) -> void* {
        void* p = wsp + off;
        off += (bytes + 255) & ~(size_t)255;
        return p;
    };
    float*    deg   = (float*)   carve(sizeof(float) * NP);
    float*    agg   = (float*)   carve(sizeof(float) * NP * CIN);
    float*    A1    = (float*)   carve(sizeof(float) * NP * CIN);
    float*    A2    = (float*)   carve(sizeof(float) * NP * CIN);
    float*    A3    = (float*)   carve(sizeof(float) * NP * CIN);
    float*    stats = (float*)   carve(sizeof(float) * 64);
    float*    aggR  = (float*)   carve(sizeof(float) * NP * 4);
    float*    aggC  = (float*)   carve(sizeof(float) * NP * 2);
    float*    aggO  = (float*)   carve(sizeof(float) * NP);
    _Float16* Wth   = (_Float16*)carve(sizeof(_Float16) * NK * CIN * CIN);
    _Float16* xh    = (_Float16*)carve(sizeof(_Float16) * NP * CIN);
    _Float16* xWh   = (_Float16*)carve(sizeof(_Float16) * NP * NK * CIN);
    (void)ws_size; (void)n_in; (void)out_size;

    const long NC = (long)N * CIN;
    dim3 blk(256);
    auto gr = [](long n) { return dim3((unsigned)((n + 255) / 256)); };

    // degree (shared by every conv)
    k_zero<<<gr(NP), blk, 0, stream>>>(deg, NP);
    k_degree<<<gr(E), blk, 0, stream>>>(ei, deg, E);

    long wmma_waves = (long)ntiles * 2;           // one wave per (tile, col-tile), k looped inside
    dim3 wmma_grid((unsigned)((wmma_waves + 7) / 8));
    dim3 edge_grid((unsigned)(((long)E + 7) / 8));

    auto run_conv = [&](const float* xin, const float* W, const float* root,
                        const float* g, const float* beta, float* yout) {
        k_to_half<<<gr(NC), blk, 0, stream>>>(xin, xh, NC);
        k_prep_w<<<gr(NK * CIN * CIN), blk, 0, stream>>>(W, Wth);
        k_zero<<<gr(NP * CIN), blk, 0, stream>>>(agg, NP * CIN);
        k_xw_wmma<<<wmma_grid, blk, 0, stream>>>(xh, Wth, xWh, ntiles);
        k_edge_msg<<<edge_grid, blk, 0, stream>>>(xWh, ei, ea, agg, E);
        k_zero<<<1, 64, 0, stream>>>(stats, 64);
        k_finalize<<<gr(NC), blk, 0, stream>>>(agg, deg, xin, root, yout, stats, N);
        k_bn_apply<<<gr(NC), blk, 0, stream>>>(yout, stats, g, beta, N);
    };

    run_conv(x,  W1, root1, g1, beta1, A1);
    run_conv(A1, W2, root2, g2, beta2, A2);
    run_conv(A1, W3, root3, g3, beta3, A3);

    // heads
    k_zero<<<gr(NP * 4), blk, 0, stream>>>(aggR, NP * 4);
    k_zero<<<gr(NP * 2), blk, 0, stream>>>(aggC, NP * 2);
    k_zero<<<gr(NP),     blk, 0, stream>>>(aggO, NP);
    k_head_edge<4><<<gr(E), blk, 0, stream>>>(A2, Wr, ei, ea, aggR, E);
    k_head_edge<2><<<gr(E), blk, 0, stream>>>(A3, Wc, ei, ea, aggC, E);
    k_head_edge<1><<<gr(E), blk, 0, stream>>>(A3, Wo, ei, ea, aggO, E);

    // output grid (zero then scatter; duplicates resolve arbitrarily, matching .at[].set)
    k_zero<<<gr(10752), blk, 0, stream>>>((float*)d_out, 10752);
    k_head_scatter<<<gr(N), blk, 0, stream>>>(aggR, aggC, aggO, deg, A2, A3,
                                              rootr, br, rootc, bc, rooto, bo,
                                              pos, batch, (float*)d_out, N);
}
